// StackLSTMParser_64922725647045
// MI455X (gfx1250) — matrix operand order repaired and checked by reading, compiled, not run on hardware
//
#include <hip/hip_runtime.h>
#include <hip/hip_bf16.h>
#include <math.h>

// ---------------------------------------------------------------------------
// StackLSTM parser for MI455X (gfx1250, wave32, WMMA bf16 16x16x32, f32 acc)
//
// Pipeline (all on `stream`):
//   1) k_convert  x11 : weights f32 -> bf16, pre-swizzled into WMMA B-fragment
//                       layout (each lane's 16 bf16 contiguous -> b128 loads)
//   2) k_compose      : tok = relu(word_emb[tokens] @ compose_W.T)  (parallel)
//   3) k_bufscan      : 64-step reverse LSTM scan, persistent 32-wave block
//   4) k_parse        : 100-step parser loop, persistent 32-wave block
// ---------------------------------------------------------------------------

typedef __attribute__((ext_vector_type(16))) __bf16 bf16x16;
typedef __attribute__((ext_vector_type(8)))  float  f32x8;

#define LDA 392            // LDS A panel stride (<=384 cols + 8 pad, bank stagger)
#define H256 256
#define B128 128

__device__ __forceinline__ unsigned short f2bf(float f) {
  unsigned u = __float_as_uint(f);
  u += 0x7FFFu + ((u >> 16) & 1u);          // round-to-nearest-even
  return (unsigned short)(u >> 16);
}
__device__ __forceinline__ float sigm(float x) { return 1.f / (1.f + __expf(-x)); }

// --- one 16x16 output tile, K = KB*32, A from LDS, B pre-swizzled in global ---
template <int KB>
__device__ __forceinline__ f32x8 gemm_tile(const unsigned short* __restrict__ Ab,
                                           int mt,
                                           const uint4* __restrict__ Bsw,
                                           int nt, f32x8 c) {
  const int lane = threadIdx.x & 31;
  const int half = lane >> 4;
  const int r    = lane & 15;
  const unsigned short* arow = Ab + (mt * 16 + r) * LDA + half * 8;
  const uint4* bp = Bsw + ((size_t)nt * KB * 32 + lane) * 2;
#pragma unroll
  for (int kb = 0; kb < KB; ++kb) {
    union { bf16x16 v; unsigned u[8]; } a;
#pragma unroll
    for (int j = 0; j < 4; ++j)               // VGPR0-3: K = kb*32 + half*8 + 2j
      a.u[j] = *(const unsigned*)(arow + kb * 32 + 2 * j);
#pragma unroll
    for (int j = 0; j < 4; ++j)               // VGPR4-7: K = kb*32 + 16 + half*8 + 2j
      a.u[4 + j] = *(const unsigned*)(arow + kb * 32 + 16 + 2 * j);
    union { bf16x16 v; uint4 q[2]; } b;
    b.q[0] = bp[kb * 64 + 0];
    b.q[1] = bp[kb * 64 + 1];
    c = __builtin_amdgcn_wmma_f32_16x16x32_bf16(false, a.v, false, b.v,
                                                (short)0, c, false, false);
  }
  return c;
}

__device__ __forceinline__ void store_z(float* __restrict__ z, int mt, int nt,
                                        const f32x8& c) {
  const int lane = threadIdx.x & 31;
  const int half = lane >> 4, cl = lane & 15;
  const int col = nt * 16 + cl;
#pragma unroll
  for (int i = 0; i < 8; ++i)
    z[(size_t)(mt * 16 + i + half * 8) * 1024 + col] = c[i];
}

// ---------------------------------------------------------------------------
// Weight conversion: B[k][n] = W[n][k] -> bf16, swizzled so that lane `l` of
// the fragment for (nt, kb) holds its 16 bf16 contiguously.
// B-fragment K map: k0 = kb*32 + (lane>=16 ? 8 : 0) + 2(j&3) + 16(j>>2)   (pairs packed LSB-first)
// ---------------------------------------------------------------------------
__global__ void k_convert(const float* __restrict__ src, int src_ld, int src_koff,
                          unsigned* __restrict__ dst, int KBtot, int N,
                          int k_lo, int k_hi) {
  int tid = blockIdx.x * blockDim.x + threadIdx.x;
  int total = (N / 16) * KBtot * 256;
  if (tid >= total) return;
  int j = tid & 7, lane = (tid >> 3) & 31, blk = tid >> 8;
  int kb = blk % KBtot, nt = blk / KBtot;
  int half = lane >> 4, cl = lane & 15;
  int k0 = kb * 32 + half * 8 + 2 * (j & 3) + 16 * (j >> 2);
  if (k0 < k_lo || k0 >= k_hi) return;
  int n = nt * 16 + cl;
  float lo = src[(size_t)n * src_ld + src_koff + (k0 - k_lo)];
  float hi = src[(size_t)n * src_ld + src_koff + (k0 - k_lo) + 1];
  dst[tid] = (unsigned)f2bf(lo) | ((unsigned)f2bf(hi) << 16);
}

// ---------------------------------------------------------------------------
// Token composition: tok(8192x128) = relu(emb_gather(8192x256) @ Bcomp)
// ---------------------------------------------------------------------------
__global__ void __launch_bounds__(256) k_compose(const int* __restrict__ tokens,
                                                 const float* __restrict__ word_emb,
                                                 const uint4* __restrict__ Bsw,
                                                 const float* __restrict__ bias,
                                                 float* __restrict__ tok_ws) {
  __shared__ unsigned short Ab[16 * LDA];
  const int mt = blockIdx.x;                       // 512 blocks, 16 rows each
  const int t = threadIdx.x;
  for (int l = t; l < 16 * 256; l += 256) {
    int r = l >> 8, c = l & 255;
    int tk = tokens[mt * 16 + r];
    Ab[r * LDA + c] = f2bf(word_emb[(size_t)tk * 256 + c]);
  }
  __syncthreads();
  const int w = t >> 5, lane = t & 31, half = lane >> 4, cl = lane & 15;
  f32x8 z8 = {0.f, 0.f, 0.f, 0.f, 0.f, 0.f, 0.f, 0.f};
  f32x8 acc = gemm_tile<8>(Ab, 0, Bsw, w, z8);     // nt = wave id (N=128 -> 8 tiles)
  const int col = w * 16 + cl;
  const float bv = bias[col];
#pragma unroll
  for (int i = 0; i < 8; ++i) {
    int grow = mt * 16 + i + half * 8;
    float v = acc[i] + bv;
    tok_ws[(size_t)grow * 128 + col] = v > 0.f ? v : 0.f;
  }
}

// ---------------------------------------------------------------------------
// pre_buffer LSTM scan over reversed tokens. Single persistent 1024-thr block.
// B matrix is fused [Wih.T ; Whh.T] (K = 128+256 = 384, KB=12).
// ---------------------------------------------------------------------------
__global__ void __launch_bounds__(1024) k_bufscan(
    const float* __restrict__ tok, const float* __restrict__ h0,
    const float* __restrict__ c0, const uint4* __restrict__ Bpb,
    const float* __restrict__ bih, const float* __restrict__ bhh,
    float* __restrict__ h_cur, float* __restrict__ c_cur,
    float* __restrict__ buf_h, float* __restrict__ z) {
  __shared__ unsigned short Ab[128 * LDA];
  const int t = threadIdx.x, w = t >> 5;
  for (int l = t; l < 128 * 256; l += 1024) {
    h_cur[l] = h0[l & 255];
    c_cur[l] = c0[l & 255];
  }
  __syncthreads();
  f32x8 z8 = {0.f, 0.f, 0.f, 0.f, 0.f, 0.f, 0.f, 0.f};
#pragma unroll 1
  for (int step = 0; step < 64; ++step) {
    const int srow = 63 - step;                    // reversed sequence
    for (int l = t; l < 128 * 128; l += 1024) {    // x segment, cols 0..127
      int r = l >> 7, c = l & 127;
      Ab[r * LDA + c] = f2bf(tok[((size_t)srow * 128 + r) * 128 + c]);
    }
    for (int l = t; l < 128 * 256; l += 1024) {    // h segment, cols 128..383
      int r = l >> 8, c = l & 255;
      Ab[r * LDA + 128 + c] = f2bf(h_cur[l]);
    }
    __syncthreads();
#pragma unroll 1
    for (int q = 0; q < 16; ++q) {                 // 512 tiles / 32 waves
      int id = w * 16 + q, mt = id >> 6, nt = id & 63;
      store_z(z, mt, nt, gemm_tile<12>(Ab, mt, Bpb, nt, z8));
    }
    __syncthreads();
    for (int l = t; l < 128 * 256; l += 1024) {    // gates
      int row = l >> 8, col = l & 255;
      const float* zr = z + (size_t)row * 1024;
      float zi = zr[col] + bih[col] + bhh[col];
      float zf = zr[256 + col] + bih[256 + col] + bhh[256 + col];
      float zg = zr[512 + col] + bih[512 + col] + bhh[512 + col];
      float zo = zr[768 + col] + bih[768 + col] + bhh[768 + col];
      float c2 = sigm(zf) * c_cur[l] + sigm(zi) * tanhf(zg);
      float h2 = sigm(zo) * tanhf(c2);
      c_cur[l] = c2;
      h_cur[l] = h2;
      buf_h[(size_t)step * 32768 + l] = h2;
    }
    __syncthreads();
  }
}

// ---------------------------------------------------------------------------
// Main parser loop: 100 sequential steps, persistent 1024-thread block.
// ---------------------------------------------------------------------------
__global__ void __launch_bounds__(1024) k_parse(
    const float* __restrict__ tok, const float* __restrict__ buf_h,
    const float* __restrict__ h0, const float* __restrict__ c0,
    float* __restrict__ stack_h, float* __restrict__ stack_c,
    float* __restrict__ ah_ws, float* __restrict__ ac_ws,
    const uint4* __restrict__ Bsum0, const uint4* __restrict__ Bsum1,
    const uint4* __restrict__ Bsum2, const float* __restrict__ sum_b,
    const uint4* __restrict__ Bact, const float* __restrict__ act_b,
    const uint4* __restrict__ Bst, const float* __restrict__ st_bih,
    const float* __restrict__ st_bhh, const uint4* __restrict__ Bh,
    const float* __restrict__ h_bih, const float* __restrict__ h_bhh,
    const float* __restrict__ action_emb, const int* __restrict__ smap,
    const int* __restrict__ bmap, float* __restrict__ z,
    float* __restrict__ out) {
  __shared__ unsigned short Ab[128 * LDA];
  __shared__ float logitsS[128 * 64];
  __shared__ int a_sel[128], pushf[128], widxS[128];
  __shared__ int sposA[128], bposA[128], sposB[128], bposB[128];
  const int t = threadIdx.x, w = t >> 5, lane = t & 31;
  const int mtw = w >> 2, g4 = w & 3, half = lane >> 4, cl = lane & 15;
  f32x8 z8 = {0.f, 0.f, 0.f, 0.f, 0.f, 0.f, 0.f, 0.f};

  for (int l = t; l < 128 * 256; l += 1024) {      // stack slot 0 + history init
    stack_h[l] = h0[l & 255];
    stack_c[l] = c0[l & 255];
    ah_ws[l] = h0[l & 255];
    ac_ws[l] = c0[l & 255];
  }
  if (t < 128) { sposA[t] = 0; bposA[t] = 64; }
  __syncthreads();

#pragma unroll 1
  for (int step = 0; step < 100; ++step) {
    // ---- summary = relu([s_state|b_state|ah] @ sum_W.T + b), 3 K=256 segs ----
    f32x8 accS[4];
#pragma unroll
    for (int q = 0; q < 4; ++q) accS[q] = z8;
    for (int l = t; l < 128 * 256; l += 1024) {    // seg0: s_state gather
      int r = l >> 8, c = l & 255, sp = sposA[r];
      Ab[r * LDA + c] = f2bf(stack_h[((size_t)sp * 128 + r) * 256 + c]);
    }
    __syncthreads();
#pragma unroll
    for (int q = 0; q < 4; ++q) accS[q] = gemm_tile<8>(Ab, mtw, Bsum0, g4 * 4 + q, accS[q]);
    __syncthreads();
    for (int l = t; l < 128 * 256; l += 1024) {    // seg1: b_state gather
      int r = l >> 8, c = l & 255, bp = bposA[r];
      float v = bp > 0 ? buf_h[((size_t)(bp - 1) * 128 + r) * 256 + c] : h0[c];
      Ab[r * LDA + c] = f2bf(v);
    }
    __syncthreads();
#pragma unroll
    for (int q = 0; q < 4; ++q) accS[q] = gemm_tile<8>(Ab, mtw, Bsum1, g4 * 4 + q, accS[q]);
    __syncthreads();
    for (int l = t; l < 128 * 256; l += 1024) {    // seg2: history state
      int r = l >> 8, c = l & 255;
      Ab[r * LDA + c] = f2bf(ah_ws[l]);
    }
    __syncthreads();
#pragma unroll
    for (int q = 0; q < 4; ++q) accS[q] = gemm_tile<8>(Ab, mtw, Bsum2, g4 * 4 + q, accS[q]);
    __syncthreads();
#pragma unroll
    for (int q = 0; q < 4; ++q) {                  // relu+bias -> Ab (A for act GEMM)
      int col = (g4 * 4 + q) * 16 + cl;
      float bb = sum_b[col];
#pragma unroll
      for (int i = 0; i < 8; ++i) {
        int row = mtw * 16 + i + half * 8;
        float v = accS[q][i] + bb;
        Ab[row * LDA + col] = f2bf(v > 0.f ? v : 0.f);
      }
    }
    __syncthreads();
    // ---- action logits (N=64): 32 tiles, one per wave ----
    {
      f32x8 a1 = gemm_tile<8>(Ab, mtw, Bact, g4, z8);
      int col = g4 * 16 + cl;
      float bb = act_b[col];
#pragma unroll
      for (int i = 0; i < 8; ++i)
        logitsS[(mtw * 16 + i + half * 8) * 64 + col] = a1[i] + bb;
    }
    __syncthreads();
    // ---- log-softmax / argmax / transition masks (1 thread per batch row) ----
    if (t < 128) {
      const float* lg = &logitsS[t * 64];
      float m = lg[0];
      int am = 0;
      for (int a = 1; a < 64; ++a)
        if (lg[a] > m) { m = lg[a]; am = a; }
      float s = 0.f;
      for (int a = 0; a < 64; ++a) s += __expf(lg[a] - m);
      float lse = m + __logf(s);
      float* op = &out[((size_t)step * 128 + t) * 64];
      for (int a = 0; a < 64; ++a) op[a] = lg[a] - lse;
      int sop = smap[am], bop = bmap[am];
      int sp = sposA[t], bp = bposA[t];
      if (sp == 0 && sop == -1) sop = 0;
      if (sp >= 40 && sop == 1) sop = 0;
      if (bp == 0 && bop == -1) bop = 0;
      a_sel[t] = am;
      pushf[t] = (sop == 1);
      int wi = sp + 1; if (wi > 40) wi = 40;
      widxS[t] = wi;
      sposB[t] = sp + sop;
      bposB[t] = bp + bop;
    }
    __syncthreads();
    // ---- stack LSTM: A=[s_in | sh[spos]], fused B (K=384) -> z ----
    for (int l = t; l < 128 * 128; l += 1024) {
      int r = l >> 7, c = l & 127, bp = bposA[r];
      float v = bp > 0 ? tok[((size_t)(bp - 1) * 128 + r) * 128 + c] : 0.f;
      Ab[r * LDA + c] = f2bf(v);
    }
    for (int l = t; l < 128 * 256; l += 1024) {
      int r = l >> 8, c = l & 255, sp = sposA[r];
      Ab[r * LDA + 128 + c] = f2bf(stack_h[((size_t)sp * 128 + r) * 256 + c]);
    }
    __syncthreads();
#pragma unroll 1
    for (int q = 0; q < 16; ++q) {
      int id = w * 16 + q, mt = id >> 6, nt = id & 63;
      store_z(z, mt, nt, gemm_tile<12>(Ab, mt, Bst, nt, z8));
    }
    __syncthreads();
    for (int l = t; l < 128 * 256; l += 1024) {    // gates + conditional push
      int row = l >> 8, col = l & 255;
      const float* zr = z + (size_t)row * 1024;
      float zi = zr[col] + st_bih[col] + st_bhh[col];
      float zf = zr[256 + col] + st_bih[256 + col] + st_bhh[256 + col];
      float zg = zr[512 + col] + st_bih[512 + col] + st_bhh[512 + col];
      float zo = zr[768 + col] + st_bih[768 + col] + st_bhh[768 + col];
      float cp = stack_c[((size_t)sposA[row] * 128 + row) * 256 + col];
      float c2 = sigm(zf) * cp + sigm(zi) * tanhf(zg);
      float h2 = sigm(zo) * tanhf(c2);
      if (pushf[row]) {
        size_t idx = ((size_t)widxS[row] * 128 + row) * 256 + col;
        stack_h[idx] = h2;
        stack_c[idx] = c2;
      }
    }
    __syncthreads();
    // ---- history LSTM: A=[action_emb[a] | ah], fused B (K=320) -> z ----
    for (int l = t; l < 128 * 64; l += 1024) {
      int r = l >> 6, c = l & 63, a = a_sel[r];
      Ab[r * LDA + c] = f2bf(action_emb[a * 64 + c]);
    }
    for (int l = t; l < 128 * 256; l += 1024) {
      int r = l >> 8, c = l & 255;
      Ab[r * LDA + 64 + c] = f2bf(ah_ws[l]);
    }
    __syncthreads();
#pragma unroll 1
    for (int q = 0; q < 16; ++q) {
      int id = w * 16 + q, mt = id >> 6, nt = id & 63;
      store_z(z, mt, nt, gemm_tile<10>(Ab, mt, Bh, nt, z8));
    }
    __syncthreads();
    for (int l = t; l < 128 * 256; l += 1024) {
      int row = l >> 8, col = l & 255;
      const float* zr = z + (size_t)row * 1024;
      float zi = zr[col] + h_bih[col] + h_bhh[col];
      float zf = zr[256 + col] + h_bih[256 + col] + h_bhh[256 + col];
      float zg = zr[512 + col] + h_bih[512 + col] + h_bhh[512 + col];
      float zo = zr[768 + col] + h_bih[768 + col] + h_bhh[768 + col];
      float c2 = sigm(zf) * ac_ws[l] + sigm(zi) * tanhf(zg);
      float h2 = sigm(zo) * tanhf(c2);
      ac_ws[l] = c2;
      ah_ws[l] = h2;
    }
    __syncthreads();
    if (t < 128) { sposA[t] = sposB[t]; bposA[t] = bposB[t]; }
    __syncthreads();
  }
}

// ---------------------------------------------------------------------------
extern "C" void kernel_launch(void* const* d_in, const int* in_sizes, int n_in,
                              void* d_out, int out_size, void* d_ws, size_t ws_size,
                              hipStream_t stream) {
  const int*   tokens     = (const int*)d_in[0];
  const float* word_emb   = (const float*)d_in[1];
  const float* compose_W  = (const float*)d_in[2];
  const float* compose_b  = (const float*)d_in[3];
  const float* h0         = (const float*)d_in[4];
  const float* c0         = (const float*)d_in[5];
  const float* pb_Wih     = (const float*)d_in[6];
  const float* pb_Whh     = (const float*)d_in[7];
  const float* pb_bih     = (const float*)d_in[8];
  const float* pb_bhh     = (const float*)d_in[9];
  const float* st_Wih     = (const float*)d_in[10];
  const float* st_Whh     = (const float*)d_in[11];
  const float* st_bih     = (const float*)d_in[12];
  const float* st_bhh     = (const float*)d_in[13];
  const float* hist_Wih   = (const float*)d_in[14];
  const float* hist_Whh   = (const float*)d_in[15];
  const float* hist_bih   = (const float*)d_in[16];
  const float* hist_bhh   = (const float*)d_in[17];
  const float* sum_W      = (const float*)d_in[18];
  const float* sum_b      = (const float*)d_in[19];
  const float* act_W      = (const float*)d_in[20];
  const float* act_b      = (const float*)d_in[21];
  const float* action_emb = (const float*)d_in[22];
  const int*   smap       = (const int*)d_in[23];
  const int*   bmap       = (const int*)d_in[24];
  float* out = (float*)d_out;

  char* ws = (char*)d_ws;
  size_t off = 0;
  auto take = [&](size_t bytes) {
    size_t r = off;
    off += (bytes + 255) & ~(size_t)255;
    return r;
  };
  unsigned* swComp = (unsigned*)(ws + take((size_t)256 * 128 * 2));  // K=256 N=128
  unsigned* swPb   = (unsigned*)(ws + take((size_t)384 * 1024 * 2)); // K=384 N=1024
  unsigned* swSt   = (unsigned*)(ws + take((size_t)384 * 1024 * 2));
  unsigned* swHi   = (unsigned*)(ws + take((size_t)320 * 1024 * 2)); // K=320 N=1024
  unsigned* swSum0 = (unsigned*)(ws + take((size_t)256 * 256 * 2));
  unsigned* swSum1 = (unsigned*)(ws + take((size_t)256 * 256 * 2));
  unsigned* swSum2 = (unsigned*)(ws + take((size_t)256 * 256 * 2));
  unsigned* swAct  = (unsigned*)(ws + take((size_t)256 * 64 * 2));
  float* tok_ws  = (float*)(ws + take((size_t)64 * 128 * 128 * 4));
  float* buf_h   = (float*)(ws + take((size_t)64 * 128 * 256 * 4));
  float* stack_h = (float*)(ws + take((size_t)41 * 128 * 256 * 4));
  float* stack_c = (float*)(ws + take((size_t)41 * 128 * 256 * 4));
  float* h_cur   = (float*)(ws + take((size_t)128 * 256 * 4));
  float* c_cur   = (float*)(ws + take((size_t)128 * 256 * 4));
  float* ah_ws   = (float*)(ws + take((size_t)128 * 256 * 4));
  float* ac_ws   = (float*)(ws + take((size_t)128 * 256 * 4));
  float* z_ws    = (float*)(ws + take((size_t)128 * 1024 * 4));

  auto conv = [&](const float* src, int ld, int koff, unsigned* dst, int KB,
                  int N, int klo, int khi) {
    int total = (N / 16) * KB * 256;
    k_convert<<<(total + 255) / 256, 256, 0, stream>>>(src, ld, koff, dst, KB, N,
                                                       klo, khi);
  };
  conv(compose_W, 256, 0, swComp, 8, 128, 0, 256);
  conv(pb_Wih, 128, 0, swPb, 12, 1024, 0, 128);
  conv(pb_Whh, 256, 0, swPb, 12, 1024, 128, 384);
  conv(st_Wih, 128, 0, swSt, 12, 1024, 0, 128);
  conv(st_Whh, 256, 0, swSt, 12, 1024, 128, 384);
  conv(hist_Wih, 64, 0, swHi, 10, 1024, 0, 64);
  conv(hist_Whh, 256, 0, swHi, 10, 1024, 64, 320);
  conv(sum_W, 768, 0,   swSum0, 8, 256, 0, 256);
  conv(sum_W, 768, 256, swSum1, 8, 256, 0, 256);
  conv(sum_W, 768, 512, swSum2, 8, 256, 0, 256);
  conv(act_W, 256, 0, swAct, 8, 64, 0, 256);

  k_compose<<<512, 256, 0, stream>>>(tokens, word_emb, (const uint4*)swComp,
                                     compose_b, tok_ws);

  k_bufscan<<<1, 1024, 0, stream>>>(tok_ws, h0, c0, (const uint4*)swPb, pb_bih,
                                    pb_bhh, h_cur, c_cur, buf_h, z_ws);

  k_parse<<<1, 1024, 0, stream>>>(
      tok_ws, buf_h, h0, c0, stack_h, stack_c, ah_ws, ac_ws,
      (const uint4*)swSum0, (const uint4*)swSum1, (const uint4*)swSum2, sum_b,
      (const uint4*)swAct, act_b, (const uint4*)swSt, st_bih, st_bhh,
      (const uint4*)swHi, hist_bih, hist_bhh, action_emb, smap, bmap, z_ws, out);
}
